// MultiHeadSelfAttention_764504179251
// MI455X (gfx1250) — matrix operand order repaired and checked by reading, not compiled, not run on hardware
//
#include <hip/hip_runtime.h>
#include <hip/hip_bf16.h>

// ---------------------------------------------------------------------------
// MI455X (gfx1250) multi-head self-attention, bf16 WMMA pipeline.
// B=1, S=4096, D=1024, H=16, DH=64.
// ---------------------------------------------------------------------------

typedef __bf16 bf16;
typedef __attribute__((ext_vector_type(16))) __bf16 v16bf;
typedef __attribute__((ext_vector_type(8)))  __bf16 v8bf;
typedef __attribute__((ext_vector_type(8)))  float  v8f;
typedef __attribute__((ext_vector_type(4)))  int    v4i;

#define SEQ 4096
#define DIM 1024
#define NH  16
#define DHEAD 64

static __device__ __forceinline__ v16bf cat8(v8bf lo, v8bf hi) {
  return __builtin_shufflevector(lo, hi, 0, 1, 2, 3, 4, 5, 6, 7,
                                         8, 9, 10, 11, 12, 13, 14, 15);
}

static __device__ __forceinline__ v8f wmma_bf16(v16bf a, v16bf b, v8f c) {
  return __builtin_amdgcn_wmma_f32_16x16x32_bf16(
      /*neg_a=*/false, a, /*neg_b=*/false, b,
      /*c_mod=*/(short)0, c, /*reuse_a=*/false, /*reuse_b=*/false);
}

// ---- CDNA5 async global->LDS copy (ASYNCcnt-tracked), 16 bytes per lane ----
static __device__ __forceinline__ void async_copy16(const bf16* g, bf16* l) {
#if __has_builtin(__builtin_amdgcn_global_load_async_to_lds_b128)
  __builtin_amdgcn_global_load_async_to_lds_b128(
      (__attribute__((address_space(1))) v4i*)(v4i*)g,
      (__attribute__((address_space(3))) v4i*)(v4i*)l,
      0, 0);
#else
  unsigned lofs = (unsigned)(uintptr_t)(__attribute__((address_space(3))) char*)l;
  asm volatile("global_load_async_to_lds_b128 %0, %1, off"
               :: "v"(lofs), "v"(g) : "memory");
#endif
}

static __device__ __forceinline__ void wait_async0() {
#if __has_builtin(__builtin_amdgcn_s_wait_asynccnt)
  __builtin_amdgcn_s_wait_asynccnt(0);
#else
  asm volatile("s_wait_asynccnt 0x0" ::: "memory");
#endif
}

// ---------------------------------------------------------------------------
// f32 -> bf16 elementwise convert
// ---------------------------------------------------------------------------
__global__ __launch_bounds__(256) void cvt_f32_bf16(const float* __restrict__ src,
                                                    bf16* __restrict__ dst, int n) {
  int i = blockIdx.x * blockDim.x + threadIdx.x;
  if (i < n) dst[i] = (bf16)src[i];
}

// ---------------------------------------------------------------------------
// C[M,N] = A[M,K] @ W[N,K]^T  with bf16 A/W, WMMA f32 accumulation.
// A block-tile (64x32) is staged through LDS with double-buffered async
// copies (shared by all 4 waves); W is streamed straight to registers.
// STORE_MODE: 0 = f32 row-major, 1 = bf16 row-major, 2 = bf16 transposed [N,M]
// Block = 128 threads (4 waves). Block tile: 64(M) x 128(N); wave tile 64x32.
// ---------------------------------------------------------------------------
template <int STORE_MODE>
__global__ __launch_bounds__(128) void gemm_xwT(const bf16* __restrict__ A,
                                                const bf16* __restrict__ W,
                                                void* __restrict__ Cout,
                                                int M, int N, int K) {
  __shared__ __align__(16) bf16 atile[2][64][32];  // 8 KiB, double-buffered

  const int tid    = threadIdx.x;
  const int lane   = tid & 31;
  const int wave   = tid >> 5;
  const int lane16 = lane & 15;
  const bool hi    = lane >= 16;
  const int  aof   = hi ? 8 : 0;   // unused; A-frags also use bof (k = 16*h + s)
  const int  bof   = hi ? 16 : 0;  // B-frag k sub-offset per lane half

  const int m0 = blockIdx.y * 64;
  const int n0 = blockIdx.x * 128 + wave * 32;

  // issue async copies for one 64x32 A tile: 256 x 16B chunks, 2 per thread
  auto issueA = [&](int k0, int buf) {
#pragma unroll
    for (int c = 0; c < 2; ++c) {
      const int chunk = tid * 2 + c;
      const int row   = chunk >> 2;
      const int col   = (chunk & 3) * 8;
      async_copy16(A + (size_t)(m0 + row) * K + k0 + col, &atile[buf][row][col]);
    }
  };

  v8f acc[4][2];
#pragma unroll
  for (int mi = 0; mi < 4; ++mi)
#pragma unroll
    for (int ni = 0; ni < 2; ++ni) acc[mi][ni] = (v8f){};

  issueA(0, 0);

  for (int k0 = 0; k0 < K; k0 += 32) {
    const int buf = (k0 >> 5) & 1;
    wait_async0();      // this wave's portion landed in LDS
    __syncthreads();    // everyone's portion landed; prev buffer free
    if (k0 + 32 < K) issueA(k0 + 32, buf ^ 1);

    v16bf af[4], bfr[2];
#pragma unroll
    for (int mi = 0; mi < 4; ++mi) {
      const bf16* ap = &atile[buf][mi * 16 + lane16][bof];
      af[mi] = cat8(*(const v8bf*)ap, *(const v8bf*)(ap + 8));
    }
#pragma unroll
    for (int ni = 0; ni < 2; ++ni) {
      const bf16* bp = W + (size_t)(n0 + ni * 16 + lane16) * K + k0 + bof;
      bfr[ni] = *(const v16bf*)bp;
      __builtin_prefetch(bp + 64, 0, 0);  // global_prefetch_b8
    }
#pragma unroll
    for (int mi = 0; mi < 4; ++mi)
#pragma unroll
      for (int ni = 0; ni < 2; ++ni)
        acc[mi][ni] = wmma_bf16(af[mi], bfr[ni], acc[mi][ni]);
  }

#pragma unroll
  for (int mi = 0; mi < 4; ++mi)
#pragma unroll
    for (int ni = 0; ni < 2; ++ni)
#pragma unroll
      for (int r = 0; r < 8; ++r) {
        const int row = m0 + mi * 16 + r + (hi ? 8 : 0);
        const int col = n0 + ni * 16 + lane16;
        const float v = acc[mi][ni][r];
        if constexpr (STORE_MODE == 0)
          ((float*)Cout)[(size_t)row * N + col] = v;
        else if constexpr (STORE_MODE == 1)
          ((bf16*)Cout)[(size_t)row * N + col] = (bf16)v;
        else
          ((bf16*)Cout)[(size_t)col * M + row] = (bf16)v;
      }
}

// ---------------------------------------------------------------------------
// Flash attention, causal. One wave owns 16 query rows of one head.
// exp2-domain online softmax; mask-free main loop + masked diagonal tail.
// ---------------------------------------------------------------------------
#define SCALE_LOG2E 0.18033688f  // (1/sqrt(64)) * log2(e)

template <bool MASK>
static __device__ __forceinline__ void flash_step(
    int kblk, int q0, int h, int lane16, bool hi, int aof, int bof,
    const bf16* __restrict__ Kc, const bf16* __restrict__ Vt,
    const v16bf (&qf)[2], float (&mrow)[8], float (&lrow)[8], v8f (&oacc)[4],
    bf16 (&pb)[16][48]) {
  // ---- scores S(16q x 32k) = Q @ K^T, two 16-key chunks ----
  v8f sc[2];
#pragma unroll
  for (int kc = 0; kc < 2; ++kc) {
    const int kb = kblk + kc * 16;
    const bf16* kp = Kc + (size_t)(kb + lane16) * DIM + h * DHEAD + bof;
    v16bf kf0 = *(const v16bf*)kp;
    v16bf kf1 = *(const v16bf*)(kp + 32);
    v8f z = (v8f){};
    sc[kc] = wmma_bf16(qf[0], kf0, z);
    sc[kc] = wmma_bf16(qf[1], kf1, sc[kc]);
  }

  // ---- online softmax in exp2 domain (rows live in 16-lane halves) ----
#pragma unroll
  for (int r = 0; r < 8; ++r) {
    float s0 = sc[0][r] * SCALE_LOG2E;
    float s1 = sc[1][r] * SCALE_LOG2E;
    if constexpr (MASK) {
      const int qrow = q0 + r + (hi ? 8 : 0);
      if (kblk + lane16 > qrow)      s0 = -__builtin_inff();
      if (kblk + 16 + lane16 > qrow) s1 = -__builtin_inff();
    }
    float mx = fmaxf(s0, s1);
#pragma unroll
    for (int d = 1; d < 16; d <<= 1) mx = fmaxf(mx, __shfl_xor(mx, d, 32));
    const float mnew = fmaxf(mrow[r], mx);
    const float corr = exp2f(mrow[r] - mnew);
    const float p0 = exp2f(s0 - mnew);
    const float p1 = exp2f(s1 - mnew);
    float ps = p0 + p1;
#pragma unroll
    for (int d = 1; d < 16; d <<= 1) ps += __shfl_xor(ps, d, 32);
    lrow[r] = lrow[r] * corr + ps;
    mrow[r] = mnew;
#pragma unroll
    for (int ni = 0; ni < 4; ++ni) oacc[ni][r] *= corr;

    const int m = r + (hi ? 8 : 0);
    pb[m][lane16]      = (bf16)p0;  // C-layout -> LDS row-major
    pb[m][16 + lane16] = (bf16)p1;
  }

  // LDS RAW fence within the wave before re-reading P in A-layout.
  asm volatile("s_wait_dscnt 0x0" ::: "memory");

  // ---- reload P as WMMA A-frag (16x32 bf16) ----
  v16bf pf = cat8(*(const v8bf*)&pb[lane16][bof],
                  *(const v8bf*)&pb[lane16][8 + bof]);

  // ---- O += P @ V : V^T rows are contiguous in keys ----
#pragma unroll
  for (int ni = 0; ni < 4; ++ni) {
    const bf16* vp = Vt + (size_t)(h * DHEAD + ni * 16 + lane16) * SEQ + kblk + bof;
    v16bf vf = *(const v16bf*)vp;
    oacc[ni] = wmma_bf16(pf, vf, oacc[ni]);
  }
}

__global__ __launch_bounds__(128) void flash_attn(const bf16* __restrict__ Q,
                                                  const bf16* __restrict__ Kc,
                                                  const bf16* __restrict__ Vt,
                                                  bf16* __restrict__ O) {
  __shared__ __align__(32) bf16 pbuf[4][16][48];  // P staging, padded rows

  const int lane   = threadIdx.x & 31;
  const int wave   = threadIdx.x >> 5;
  const int lane16 = lane & 15;
  const bool hi    = lane >= 16;
  const int  aof   = hi ? 8 : 0;
  const int  bof   = hi ? 16 : 0;

  const int h  = blockIdx.y;
  const int q0 = blockIdx.x * 64 + wave * 16;

  // Preload Q A-frags for the whole DH=64 (two k-steps of 32).
  v16bf qf[2];
#pragma unroll
  for (int kk = 0; kk < 2; ++kk) {
    const bf16* qp = Q + (size_t)(q0 + lane16) * DIM + h * DHEAD + kk * 32 + bof;
    qf[kk] = cat8(*(const v8bf*)qp, *(const v8bf*)(qp + 8));
  }

  float mrow[8], lrow[8];
  v8f oacc[4];
#pragma unroll
  for (int r = 0; r < 8; ++r) { mrow[r] = -__builtin_inff(); lrow[r] = 0.0f; }
#pragma unroll
  for (int ni = 0; ni < 4; ++ni) oacc[ni] = (v8f){};

  const int kend        = q0 + 16;   // causal upper bound (exclusive)
  const int nomask_end  = q0 & ~31;  // blocks with kblk+31 <= q0 need no mask
  int kblk = 0;
  for (; kblk < nomask_end; kblk += 32)
    flash_step<false>(kblk, q0, h, lane16, hi, aof, bof, Kc, Vt, qf,
                      mrow, lrow, oacc, pbuf[wave]);
  for (; kblk < kend; kblk += 32)
    flash_step<true>(kblk, q0, h, lane16, hi, aof, bof, Kc, Vt, qf,
                     mrow, lrow, oacc, pbuf[wave]);

  // ---- normalize and store ----
#pragma unroll
  for (int ni = 0; ni < 4; ++ni)
#pragma unroll
    for (int r = 0; r < 8; ++r) {
      const int row = q0 + r + (hi ? 8 : 0);
      const int col = h * DHEAD + ni * 16 + lane16;
      O[(size_t)row * DIM + col] = (bf16)(oacc[ni][r] / lrow[r]);
    }
}

// ---------------------------------------------------------------------------
// Host-side launcher
// ---------------------------------------------------------------------------
extern "C" void kernel_launch(void* const* d_in, const int* in_sizes, int n_in,
                              void* d_out, int out_size, void* d_ws, size_t ws_size,
                              hipStream_t stream) {
  (void)in_sizes; (void)n_in; (void)out_size; (void)ws_size;

  const float* x  = (const float*)d_in[0];
  const float* Wq = (const float*)d_in[1];
  const float* Wk = (const float*)d_in[2];
  const float* Wv = (const float*)d_in[3];
  const float* Wo = (const float*)d_in[4];
  float* out = (float*)d_out;

  const size_t XN = (size_t)SEQ * DIM;  // 4 Mi elems
  const size_t WN = (size_t)DIM * DIM;  // 1 Mi elems

  char* ws = (char*)d_ws;
  bf16* xb  = (bf16*)(ws);                      // 8 MiB
  bf16* wqb = (bf16*)(ws + (8u << 20));         // 2 MiB
  bf16* wkb = (bf16*)(ws + (10u << 20));
  bf16* wvb = (bf16*)(ws + (12u << 20));
  bf16* wob = (bf16*)(ws + (14u << 20));
  bf16* Qb  = (bf16*)(ws + (16u << 20));        // 8 MiB each
  bf16* Kb  = (bf16*)(ws + (24u << 20));
  bf16* Vtb = (bf16*)(ws + (32u << 20));        // transposed [D, S]
  bf16* Ob  = (bf16*)(ws + (40u << 20));

  // 1) convert to bf16
  cvt_f32_bf16<<<(int)(XN / 256), 256, 0, stream>>>(x, xb, (int)XN);
  cvt_f32_bf16<<<(int)(WN / 256), 256, 0, stream>>>(Wq, wqb, (int)WN);
  cvt_f32_bf16<<<(int)(WN / 256), 256, 0, stream>>>(Wk, wkb, (int)WN);
  cvt_f32_bf16<<<(int)(WN / 256), 256, 0, stream>>>(Wv, wvb, (int)WN);
  cvt_f32_bf16<<<(int)(WN / 256), 256, 0, stream>>>(Wo, wob, (int)WN);

  // 2) projections (N tiles of 128, M tiles of 64)
  dim3 ggrid(DIM / 128, SEQ / 64);
  gemm_xwT<1><<<ggrid, 128, 0, stream>>>(xb, wqb, Qb,  SEQ, DIM, DIM);
  gemm_xwT<1><<<ggrid, 128, 0, stream>>>(xb, wkb, Kb,  SEQ, DIM, DIM);
  gemm_xwT<2><<<ggrid, 128, 0, stream>>>(xb, wvb, Vtb, SEQ, DIM, DIM);  // V^T

  // 3) causal flash attention
  dim3 agrid(SEQ / 64, NH);
  flash_attn<<<agrid, 128, 0, stream>>>(Qb, Kb, Vtb, Ob);

  // 4) output projection, f32 result
  gemm_xwT<0><<<ggrid, 128, 0, stream>>>(Ob, wob, out, SEQ, DIM, DIM);
}
